// LSTM_76347338653922
// MI455X (gfx1250) — matrix operand order repaired and checked by reading, not compiled, not run on hardware
//
#include <hip/hip_runtime.h>
#include <hip/hip_bf16.h>

// ---------------- problem constants ----------------
static constexpr int Bb    = 16;     // batch
static constexpr int Tt    = 128;    // time steps
static constexpr int In    = 256;    // input feature
static constexpr int Hh    = 256;    // hidden
static constexpr int KTOT  = 768;    // [x | atten | h]
static constexpr int NWG   = 8;      // workgroups (each owns 32 H-channels -> 128 gate cols)
static constexpr int NCOLW = 128;    // gate columns per WG
static constexpr int KT    = 24;     // K tiles of 32
static constexpr int THREADS = 256;  // 8 waves (wave32); wave i owns N-tile i

// LDS layout (bytes)
static constexpr int LDS_W    = 0;                      // 192 frags * 1024B = 196608
static constexpr int LDS_A    = 196608;                 // 24 frags * 1024B  = 24576
static constexpr int LDS_Z    = 221184;                 // 16*128 f32        = 8192
static constexpr int LDS_C    = 229376;                 // 16*32 f32         = 2048
static constexpr int LDS_BIAS = 231424;                 // 128 f32           = 512
static constexpr int LDS_BYTES = 231936;

// ws layout
static constexpr int WS_CTR  = 0;      // 129+ u32 barrier counters
static constexpr int WS_H    = 1024;   // 16*256 f32 hidden
static constexpr int WS_ATT  = 17408;  // 16*256 f32 atten running sum

typedef __attribute__((ext_vector_type(16))) __bf16 v16bf;
typedef __attribute__((ext_vector_type(8)))  float  v8f;
typedef __attribute__((ext_vector_type(4)))  float  v4f;

__device__ __forceinline__ unsigned short f2bf(float x) {
  unsigned int u = __float_as_uint(x);
  unsigned int r = u + 0x7FFFu + ((u >> 16) & 1u);   // round-to-nearest-even
  return (unsigned short)(r >> 16);
}
__device__ __forceinline__ unsigned long long pack4bf(v4f v) {
  return  (unsigned long long)f2bf(v.x)
       | ((unsigned long long)f2bf(v.y) << 16)
       | ((unsigned long long)f2bf(v.z) << 32)
       | ((unsigned long long)f2bf(v.w) << 48);
}

// CDNA5 hardware V_TANH_F32 transcendental.
__device__ __forceinline__ float fast_tanh(float x) {
#if __has_builtin(__builtin_amdgcn_tanhf)
  return __builtin_amdgcn_tanhf(x);
#elif __has_builtin(__builtin_amdgcn_tanh_f32)
  return __builtin_amdgcn_tanh_f32(x);
#else
  float e = __expf(2.0f * x);
  return 1.0f - 2.0f / (e + 1.0f);   // stable, saturates at +/-1
#endif
}
__device__ __forceinline__ float sigm(float x) {
  return 0.5f * fast_tanh(0.5f * x) + 0.5f;          // sigmoid via tanh (1 TRANS op)
}

__device__ __forceinline__ void grid_barrier(unsigned int* c) {
  __threadfence();
  __syncthreads();
  if (threadIdx.x == 0) {
    __hip_atomic_fetch_add(c, 1u, __ATOMIC_ACQ_REL, __HIP_MEMORY_SCOPE_AGENT);
    while (__hip_atomic_load(c, __ATOMIC_ACQUIRE, __HIP_MEMORY_SCOPE_AGENT) < (unsigned)NWG) {
      __builtin_amdgcn_s_sleep(1);
    }
  }
  __syncthreads();
  __threadfence();
}

__global__ void LSTM_zero_ctr(unsigned int* c, int n) {
  int i = threadIdx.x + blockIdx.x * blockDim.x;
  if (i < n) c[i] = 0u;
}

__global__ __launch_bounds__(THREADS, 1)
void LSTM_persistent(const float* __restrict__ x,
                     const float* __restrict__ h0,
                     const float* __restrict__ c0,
                     const float* __restrict__ out0,
                     const float* __restrict__ Wih,   // (512,1024)
                     const float* __restrict__ Whh,   // (256,1024)
                     const float* __restrict__ bih,
                     const float* __restrict__ bhh,
                     float* __restrict__ dout,        // 6 * 4096
                     unsigned int* __restrict__ ctr,
                     float* __restrict__ hbuf,        // (16,256)
                     float* __restrict__ abuf)        // (16,256) running atten sum
{
  extern __shared__ char lds[];
  unsigned short* ldsW    = (unsigned short*)(lds + LDS_W);
  unsigned short* ldsA    = (unsigned short*)(lds + LDS_A);
  float*          ldsZ    = (float*)(lds + LDS_Z);
  float*          ldsC    = (float*)(lds + LDS_C);
  float*          ldsBias = (float*)(lds + LDS_BIAS);

  const int tid  = threadIdx.x;
  const int wg   = blockIdx.x;       // 0..7
  const int lane = tid & 31;
  const int wv   = tid >> 5;         // wave id == N-tile

  // ---- one-time: weights -> LDS as WMMA bf16 B-fragments (contiguous k = 16*(lane/16) + i order) ----
  // WG's gate columns: local n in [0,128): global col gc = 256*(n/32) + 32*wg + (n%32)
  for (int e = tid; e < KTOT * NCOLW; e += THREADS) {
    int K  = e >> 7;            // 0..767 (row of stacked [Wih;Whh])
    int n  = e & 127;
    int gc = ((n >> 5) << 8) + (wg << 5) + (n & 31);
    float w = (K < 512) ? Wih[K * 1024 + gc] : Whh[(K - 512) * 1024 + gc];
    int kt = K >> 5, kk = K & 31;
    int ntl = n >> 4;
    int l = (n & 15) + ((kk >> 4) << 4);   // lane: col = l%16, K-half group = l/16
    int i = kk & 15;                       // 16 consecutive K per lane
    ldsW[((ntl * KT + kt) << 9) + (l << 4) + i] = f2bf(w);
  }
  for (int n = tid; n < NCOLW; n += THREADS) {
    int gc = ((n >> 5) << 8) + (wg << 5) + (n & 31);
    ldsBias[n] = bih[gc] + bhh[gc];
  }
  // ---- per-WG state init ----
  for (int e = tid; e < Bb * 32; e += THREADS) {
    int b = e >> 5, j = e & 31;
    int g = b * Hh + (wg << 5) + j;
    ldsC[e] = c0[g];
    hbuf[g] = h0[g];       // initial hidden
    abuf[g] = out0[g];     // buf row 0 seeds the attention running sum
  }
  grid_barrier(ctr + 0);

  for (int t = 0; t < Tt; ++t) {
    // ---- build A = [x_t | atten | h] (16x768) as bf16 A-fragments in LDS ----
    // Vectorized: 4-aligned K groups keep lane fixed and half-indices consecutive,
    // so each group is one global b128 load + one ds b64 store.
    #pragma unroll 2
    for (int e = tid; e < (Bb * KTOT) / 4; e += THREADS) {
      int m  = e / (KTOT / 4);             // 192 vec4 per batch row
      int K  = (e - m * (KTOT / 4)) << 2;  // 4-aligned K
      v4f v;
      if (K < In) {
        v = *(const v4f*)&x[(m * Tt + t) * In + K];
        if ((t + 1 < Tt) && ((K & 31) == 0))   // one prefetch per 128B line
          __builtin_prefetch(&x[(m * Tt + t + 1) * In + K], 0, 0);
      } else if (K < 512) {
        v = *(const v4f*)&abuf[m * Hh + (K - In)];
      } else {
        v = *(const v4f*)&hbuf[m * Hh + (K - 512)];
      }
      int kt = K >> 5, kk = K & 31;
      int l = m + ((kk >> 4) << 4);                   // lane (constant over the 4)
      int i = kk & 15;                                // half index (4-aligned)
      *(unsigned long long*)&ldsA[(kt << 9) + (l << 4) + i] = pack4bf(v);
    }
    __syncthreads();

    // ---- GEMM: wave wv computes its 16x16 tile, K = 24 x (16x16x32 bf16 WMMA) ----
    {
      float bv = ldsBias[(wv << 4) + (lane & 15)];    // C layout: col = lane%16
      v8f acc = (v8f){bv, bv, bv, bv, bv, bv, bv, bv};
      const v16bf* Afr = (const v16bf*)ldsA;
      const v16bf* Bfr = (const v16bf*)(ldsW + ((wv * KT) << 9));
      #pragma unroll 4
      for (int kt = 0; kt < KT; ++kt) {
        v16bf a = Afr[(kt << 5) + lane];
        v16bf b = Bfr[(kt << 5) + lane];
        acc = __builtin_amdgcn_wmma_f32_16x16x32_bf16(false, a, false, b,
                                                      (short)0, acc, false, false);
      }
      int n    = (wv << 4) + (lane & 15);
      int mrow = (lane >> 4) << 3;                    // rows M = v + 8*(lane/16)
      #pragma unroll
      for (int v = 0; v < 8; ++v)
        ldsZ[(mrow + v) * NCOLW + n] = acc[v];
    }
    __syncthreads();

    // ---- gates + state update; softmax over singleton axis == 1 => atten = S ----
    #pragma unroll
    for (int r = 0; r < 2; ++r) {
      int e = tid + (r << 8);                         // 0..511
      int b = e >> 5, j = e & 31;
      float zi = ldsZ[b * NCOLW + j];
      float zf = ldsZ[b * NCOLW + 32 + j];
      float zg = ldsZ[b * NCOLW + 64 + j];
      float zo = ldsZ[b * NCOLW + 96 + j];
      float c  = ldsC[e];
      float cn = sigm(zf) * c + sigm(zi) * fast_tanh(zg);
      float hn = sigm(zo) * fast_tanh(cn);
      ldsC[e] = cn;
      int g = b * Hh + (wg << 5) + j;
      float an = abuf[g] + hn;                        // S += h_new
      if (t == Tt - 2) {                              // h_prevs[-1], c_prevs[-1], attens[-1]
        dout[3 * 4096 + g] = hn;
        dout[4 * 4096 + g] = cn;
        dout[5 * 4096 + g] = an;
      }
      if (t == Tt - 1) {                              // out_f, h_f, c_f
        dout[g]            = hn;
        dout[4096 + g]     = hn;
        dout[2 * 4096 + g] = cn;
      }
      hbuf[g] = hn;
      abuf[g] = an;
    }
    grid_barrier(ctr + t + 1);
  }
}

extern "C" void kernel_launch(void* const* d_in, const int* in_sizes, int n_in,
                              void* d_out, int out_size, void* d_ws, size_t ws_size,
                              hipStream_t stream) {
  (void)in_sizes; (void)n_in; (void)out_size; (void)ws_size;
  const float* x    = (const float*)d_in[0];
  const float* h0   = (const float*)d_in[1];
  const float* c0   = (const float*)d_in[2];
  const float* out0 = (const float*)d_in[3];
  const float* Wih  = (const float*)d_in[4];
  const float* Whh  = (const float*)d_in[5];
  const float* bih  = (const float*)d_in[6];
  const float* bhh  = (const float*)d_in[7];
  // MLP weights d_in[8..13] are numerically dead (softmax over singleton axis == 1).

  float*        dout = (float*)d_out;
  unsigned int* ctr  = (unsigned int*)((char*)d_ws + WS_CTR);
  float*        hbuf = (float*)((char*)d_ws + WS_H);
  float*        abuf = (float*)((char*)d_ws + WS_ATT);

  (void)hipFuncSetAttribute((const void*)LSTM_persistent,
                            hipFuncAttributeMaxDynamicSharedMemorySize, LDS_BYTES);

  LSTM_zero_ctr<<<1, 256, 0, stream>>>(ctr, 256);
  LSTM_persistent<<<NWG, THREADS, LDS_BYTES, stream>>>(
      x, h0, c0, out0, Wih, Whh, bih, bhh, dout, ctr, hbuf, abuf);
}
